// MistralAttention_16303695855786
// MI455X (gfx1250) — matrix operand order repaired and checked by reading, not compiled, not run on hardware
//
#include <hip/hip_runtime.h>

// ---------------------------------------------------------------------------
// Types matching gfx1250 WMMA operand layouts (wave32)
// ---------------------------------------------------------------------------
typedef _Float16 half_t;
typedef __attribute__((ext_vector_type(8)))  _Float16 v8h;
typedef __attribute__((ext_vector_type(16))) _Float16 v16h;
typedef __attribute__((ext_vector_type(4)))  float    v4f;
typedef __attribute__((ext_vector_type(8)))  float    v8f;
typedef __attribute__((ext_vector_type(4)))  unsigned u32x4;
typedef __attribute__((ext_vector_type(4)))  int      i32x4;
typedef __attribute__((ext_vector_type(8)))  int      i32x8;

static __device__ __forceinline__ v16h cat8(v8h lo, v8h hi) {
    union { v16h v; v8h h[2]; } u;
    u.h[0] = lo; u.h[1] = hi;
    return u.v;
}
static __device__ __forceinline__ v8f zero8() {
    v8f z;
#pragma unroll
    for (int i = 0; i < 8; ++i) z[i] = 0.0f;
    return z;
}
static __device__ __forceinline__ i32x4 zero4i() {
    i32x4 z;
#pragma unroll
    for (int i = 0; i < 4; ++i) z[i] = 0;
    return z;
}
static __device__ __forceinline__ i32x8 zero8i() {
    i32x8 z;
#pragma unroll
    for (int i = 0; i < 8; ++i) z[i] = 0;
    return z;
}
static __device__ __forceinline__ v8f wmma_f16(v16h a, v16h b, v8f c) {
    return __builtin_amdgcn_wmma_f32_16x16x32_f16(
        false, a, false, b, (short)0, c, false, false);
}

// Problem constants
static constexpr int Bn   = 2;
static constexpr int S    = 2048;
static constexpr int E    = 4096;
static constexpr int H    = 32;
static constexpr int KVH  = 8;
static constexpr int D    = 128;
static constexpr int Mrow = Bn * S;       // 4096 flattened rows

// ---------------------------------------------------------------------------
// Weight transpose + f32->f16 convert:  WT[n*K + k] = (f16) W[k*N + n]
// ---------------------------------------------------------------------------
__global__ __launch_bounds__(256) void transpose_convert_kernel(
    const float* __restrict__ W, half_t* __restrict__ WT, int K, int N)
{
    __shared__ half_t tile[32][33];
    const int nb = blockIdx.x * 32;
    const int kb = blockIdx.y * 32;
    const int tx = threadIdx.x & 31;
    const int ty = threadIdx.x >> 5;          // 0..7
#pragma unroll
    for (int i = 0; i < 32; i += 8)
        tile[ty + i][tx] = (half_t)W[(size_t)(kb + ty + i) * N + nb + tx];
    __syncthreads();
#pragma unroll
    for (int i = 0; i < 32; i += 8)
        WT[(size_t)(nb + ty + i) * K + kb + tx] = tile[tx][ty + i];
}

// ---------------------------------------------------------------------------
// Tiled WMMA GEMM:  C[M,N] = A[M,K] * B[K,N], B given pre-transposed f16
// BT[N,K]. B-fragments load straight from global (contiguous per the ISA
// 16-bit B layout). A tile staged in LDS: f32 A via cooperative convert,
// f16 A via the Tensor Data Mover (tensor_load_to_lds, TENSORcnt).
// 8 waves: wave w owns rows [w*16, w*16+16), both 16-col sub-tiles of BN=32.
// ---------------------------------------------------------------------------
template <typename TA, bool OUT_F32>
__global__ __launch_bounds__(256) void gemm_wmma_kernel(
    const TA* __restrict__ A, const half_t* __restrict__ BT,
    half_t* __restrict__ Ch, float* __restrict__ Cf,
    int M, int N, int K)
{
    constexpr int BM = 128, BN = 32, BK = 64;
    __shared__ half_t sA[BM * BK];            // row-major A tile (f16), 16 KB

    const int tid  = threadIdx.x;
    const int wave = tid >> 5;
    const int lane = tid & 31;
    const int m16  = lane & 15;
    const int hi   = lane >> 4;

    const int rowBase = blockIdx.y * BM;
    const int colBase = blockIdx.x * BN;

    v8f acc0 = zero8();
    v8f acc1 = zero8();

    // B row pointers (transposed weights: row n holds K contiguous halves)
    const half_t* bp0 = BT + (size_t)(colBase + m16) * K;
    const half_t* bp1 = BT + (size_t)(colBase + 16 + m16) * K;

    const int arow   = tid >> 1;              // 0..127
    const int achunk = (tid & 1) * 32;        // 0 or 32

    for (int kb = 0; kb < K; kb += BK) {
        // ---- stage A tile into LDS ----
        if constexpr (__is_same(TA, float)) {
            const v4f* ap4 = (const v4f*)(A + (size_t)(rowBase + arow) * K + kb + achunk);
            half_t tmp[32];
#pragma unroll
            for (int j = 0; j < 8; ++j) {
                v4f f = ap4[j];
#pragma unroll
                for (int e = 0; e < 4; ++e) tmp[j * 4 + e] = (half_t)f[e];
            }
#pragma unroll
            for (int j = 0; j < 4; ++j)
                *(v8h*)&sA[arow * BK + achunk + j * 8] = *(v8h*)&tmp[j * 8];
        } else {
            // Tensor Data Mover: one descriptor copies the whole 128x64-half
            // tile (row stride K halves) global -> LDS. Issued by wave 0.
            if (tid < 32) {
                const unsigned long long gaddr =
                    (unsigned long long)(uintptr_t)(A + (size_t)rowBase * K + kb);
                const unsigned ldsAddr = (unsigned)(uintptr_t)&sA[0];
                u32x4 g0;
                g0[0] = 1u;                                   // count=1 (valid D#)
                g0[1] = ldsAddr;                              // lds_addr
                g0[2] = (unsigned)(gaddr & 0xffffffffull);    // global_addr lo
                g0[3] = (unsigned)((gaddr >> 32) & 0x1ffffffull) | 0x80000000u; // hi | type=2
                i32x8 g1;
                g1[0] = 1 << 16;                              // data_size = 2 bytes
                g1[1] = (int)((K & 0xffff) << 16);            // tensor_dim0 lo16
                g1[2] = (int)((K >> 16) & 0xffff) |
                        (int)((M & 0xffff) << 16);            // dim0 hi | tensor_dim1 lo
                g1[3] = (int)((M >> 16) & 0xffff) |
                        (BK << 16);                           // dim1 hi | tile_dim0
                g1[4] = BM;                                   // tile_dim1 (tile_dim2=0)
                g1[5] = K;                                    // tensor_dim0_stride lo32
                g1[6] = 0;                                    // stride hi | dim1_stride lo
                g1[7] = 0;
                __builtin_amdgcn_tensor_load_to_lds(g0, g1, zero4i(), zero4i(),
                                                    zero8i(), 0);
                __builtin_amdgcn_s_wait_tensorcnt(0);
            }
        }
        __syncthreads();

        // ---- prefetch next k-tile of B into L2 ----
        if (kb + BK < K) {
            __builtin_prefetch(bp0 + kb + BK, 0, 1);
            __builtin_prefetch(bp1 + kb + BK, 0, 1);
        }

        // ---- fragments + WMMA (2 k-steps of 32) ----
        const int am = wave * 16 + m16;
#pragma unroll
        for (int ks = 0; ks < 2; ++ks) {
            v16h af = cat8(*(v8h*)&sA[am * BK + ks * 32 + hi * 8],
                           *(v8h*)&sA[am * BK + ks * 32 + 16 + hi * 8]);
            v16h bf0 = cat8(*(const v8h*)(bp0 + kb + ks * 32 + hi * 8),
                            *(const v8h*)(bp0 + kb + ks * 32 + 16 + hi * 8));
            v16h bf1 = cat8(*(const v8h*)(bp1 + kb + ks * 32 + hi * 8),
                            *(const v8h*)(bp1 + kb + ks * 32 + 16 + hi * 8));
            acc0 = wmma_f16(af, bf0, acc0);
            acc1 = wmma_f16(af, bf1, acc1);
        }
        __syncthreads();
    }

    // ---- epilogue: C/D layout row = r + 8*hi, col = lane%16 ----
#pragma unroll
    for (int r = 0; r < 8; ++r) {
        const size_t grow = (size_t)(rowBase + wave * 16 + r + hi * 8);
        const int gcol0 = colBase + m16;
        const int gcol1 = colBase + 16 + m16;
        if constexpr (OUT_F32) {
            Cf[grow * N + gcol0] = acc0[r];
            Cf[grow * N + gcol1] = acc1[r];
        } else {
            Ch[grow * N + gcol0] = (half_t)acc0[r];
            Ch[grow * N + gcol1] = (half_t)acc1[r];
        }
    }
}

// ---------------------------------------------------------------------------
// RoPE applied in-place to f16 Q (B,S,H,D) and K (B,S,KVH,D).
// position_ids == arange(S). One thread per rotation pair (d, d+64).
// ---------------------------------------------------------------------------
__global__ __launch_bounds__(256) void rope_kernel(half_t* __restrict__ Qp,
                                                   half_t* __restrict__ Kp)
{
    int idx = blockIdx.x * 256 + threadIdx.x;
    const int d    = idx & 63;  idx >>= 6;
    const int head = idx % (H + KVH);  idx /= (H + KVH);
    const int s    = idx % S;
    const int b    = idx / S;

    const float inv_freq = __powf(10000.0f, -(float)d * (1.0f / 64.0f));
    float sn, cs;
    __sincosf((float)s * inv_freq, &sn, &cs);

    half_t* base = (head < H)
        ? Qp + ((size_t)(b * S + s) * H + head) * D
        : Kp + ((size_t)(b * S + s) * KVH + (head - H)) * D;
    const float x1 = (float)base[d];
    const float x2 = (float)base[d + 64];
    base[d]      = (half_t)(x1 * cs - x2 * sn);
    base[d + 64] = (half_t)(x2 * cs + x1 * sn);
}

// ---------------------------------------------------------------------------
// Flash attention (causal, GQA 4:1). One block = one (b,h) pair x 64 q-rows,
// 4 waves x 16 rows. Streams 32-key tiles with online softmax.
// QK^T and P*V both via v_wmma_f32_16x16x32_f16 (16 WMMAs per k-tile).
// ---------------------------------------------------------------------------
__global__ __launch_bounds__(128) void flash_attn_kernel(
    const half_t* __restrict__ Q, const half_t* __restrict__ Kp,
    const half_t* __restrict__ Vp, half_t* __restrict__ Oh)
{
    constexpr int QT = 64, KT = 32;
    __shared__ half_t sVT[D * KT];       // V tile transposed: [d*32 + key] (8 KB)
    __shared__ half_t sP[4][16 * KT];    // per-wave P tile 16x32 (4 KB)

    const int tid  = threadIdx.x;
    const int wave = tid >> 5;
    const int lane = tid & 31;
    const int m16  = lane & 15;
    const int hi   = lane >> 4;

    const int ntile = S / QT;                       // 32
    const int tile  = blockIdx.x % ntile;
    const int bh    = blockIdx.x / ntile;
    const int b     = bh / H;
    const int h     = bh % H;
    const int kvh   = h / (H / KVH);

    // Preload Q fragments (16 rows x 128, 4 k-steps of 32)
    const int qrow = tile * QT + wave * 16 + m16;
    const half_t* qp = Q + ((size_t)(b * S + qrow) * H + h) * D;
    v16h qf[4];
#pragma unroll
    for (int s = 0; s < 4; ++s)
        qf[s] = cat8(*(const v8h*)(qp + s * 32 + hi * 8),
                     *(const v8h*)(qp + s * 32 + 16 + hi * 8));

    float mrow[8], lrow[8];
    v8f og[8];
#pragma unroll
    for (int r = 0; r < 8; ++r) { mrow[r] = -3.0e38f; lrow[r] = 0.0f; }
#pragma unroll
    for (int g = 0; g < 8; ++g) og[g] = zero8();

    const int vkey = tid & (KT - 1);
    const int vdc  = (tid >> 5) * 32;
    const int nkt  = (tile * QT + QT) / KT;          // causal bound
    const float scale = 0.08838834764831845f;        // 1/sqrt(128)

    for (int kt = 0; kt < nkt; ++kt) {
        const int kb = kt * KT;
        // ---- stage V tile transposed into LDS ----
        {
            const half_t* vp = Vp + ((size_t)(b * S + kb + vkey) * KVH + kvh) * D + vdc;
            half_t tmp[32];
#pragma unroll
            for (int j = 0; j < 4; ++j) *(v8h*)&tmp[j * 8] = ((const v8h*)vp)[j];
#pragma unroll
            for (int j = 0; j < 32; ++j) sVT[(vdc + j) * KT + vkey] = tmp[j];
        }
        __syncthreads();

        // ---- scores: S = Q * K^T (B-frags contiguous from row-major K) ----
        v8f s0 = zero8(), s1 = zero8();
        {
            const half_t* kp0 = Kp + ((size_t)(b * S + kb + m16) * KVH + kvh) * D;
            const half_t* kp1 = Kp + ((size_t)(b * S + kb + 16 + m16) * KVH + kvh) * D;
#pragma unroll
            for (int s = 0; s < 4; ++s) {
                v16h b0 = cat8(*(const v8h*)(kp0 + s * 32 + hi * 8),
                               *(const v8h*)(kp0 + s * 32 + 16 + hi * 8));
                v16h b1 = cat8(*(const v8h*)(kp1 + s * 32 + hi * 8),
                               *(const v8h*)(kp1 + s * 32 + 16 + hi * 8));
                s0 = wmma_f16(qf[s], b0, s0);
                s1 = wmma_f16(qf[s], b1, s1);
            }
        }

        // ---- online softmax (rows live in 16-lane halves; reduce via shfl) ----
#pragma unroll
        for (int r = 0; r < 8; ++r) {
            const int row = tile * QT + wave * 16 + r + hi * 8;
            float f0 = s0[r] * scale;
            float f1 = s1[r] * scale;
            if (kb + m16 > row)      f0 = -1.0e9f;
            if (kb + 16 + m16 > row) f1 = -1.0e9f;
            float mt = fmaxf(f0, f1);
#pragma unroll
            for (int o = 1; o < 16; o <<= 1) mt = fmaxf(mt, __shfl_xor(mt, o));
            const float mnew  = fmaxf(mrow[r], mt);
            const float alpha = __expf(mrow[r] - mnew);
            const float p0 = __expf(f0 - mnew);
            const float p1 = __expf(f1 - mnew);
            float sum = p0 + p1;
#pragma unroll
            for (int o = 1; o < 16; o <<= 1) sum += __shfl_xor(sum, o);
            lrow[r] = lrow[r] * alpha + sum;
            mrow[r] = mnew;
#pragma unroll
            for (int g = 0; g < 8; ++g) og[g][r] *= alpha;
            sP[wave][(r + hi * 8) * KT + m16]      = (half_t)p0;
            sP[wave][(r + hi * 8) * KT + 16 + m16] = (half_t)p1;
        }

        // ---- O += P * V ----
        v16h pf = cat8(*(v8h*)&sP[wave][m16 * KT + hi * 8],
                       *(v8h*)&sP[wave][m16 * KT + 16 + hi * 8]);
#pragma unroll
        for (int g = 0; g < 8; ++g) {
            v16h vf = cat8(*(v8h*)&sVT[(g * 16 + m16) * KT + hi * 8],
                           *(v8h*)&sVT[(g * 16 + m16) * KT + 16 + hi * 8]);
            og[g] = wmma_f16(pf, vf, og[g]);
        }
        __syncthreads();
    }

    // ---- normalize + write (b, s, h*D) layout for the out-proj GEMM ----
#pragma unroll
    for (int r = 0; r < 8; ++r) {
        const int row = tile * QT + wave * 16 + r + hi * 8;
        const float inv = 1.0f / lrow[r];
        half_t* op = Oh + ((size_t)(b * S + row) * H + h) * D;
#pragma unroll
        for (int g = 0; g < 8; ++g)
            op[g * 16 + m16] = (half_t)(og[g][r] * inv);
    }
}

// ---------------------------------------------------------------------------
// Launcher
// ---------------------------------------------------------------------------
extern "C" void kernel_launch(void* const* d_in, const int* in_sizes, int n_in,
                              void* d_out, int out_size, void* d_ws, size_t ws_size,
                              hipStream_t stream)
{
    (void)in_sizes; (void)n_in; (void)out_size; (void)ws_size;
    const float* query = (const float*)d_in[0];
    const float* key   = (const float*)d_in[1];
    const float* value = (const float*)d_in[2];
    // d_in[3] = position_ids (arange(S)), d_in[4] = causal mask: both implicit.
    const float* Wq = (const float*)d_in[5];
    const float* Wk = (const float*)d_in[6];
    const float* Wv = (const float*)d_in[7];
    const float* Wo = (const float*)d_in[8];
    float* out = (float*)d_out;

    half_t* ws  = (half_t*)d_ws;
    half_t* WqT = ws;                                  // (H*D) x E
    half_t* WkT = WqT + (size_t)E * H * D;             // (KVH*D) x E
    half_t* WvT = WkT + (size_t)E * KVH * D;
    half_t* WoT = WvT + (size_t)E * KVH * D;           // E x (H*D)
    half_t* Qp  = WoT + (size_t)H * D * E;
    half_t* Kp  = Qp  + (size_t)Mrow * H * D;
    half_t* Vp  = Kp  + (size_t)Mrow * KVH * D;
    half_t* AO  = Vp  + (size_t)Mrow * KVH * D;

    const dim3 blk256(256);

    // Weight transpose + convert (W[K,N] f32 -> WT[N,K] f16)
    transpose_convert_kernel<<<dim3((H * D) / 32, E / 32), blk256, 0, stream>>>(Wq, WqT, E, H * D);
    transpose_convert_kernel<<<dim3((KVH * D) / 32, E / 32), blk256, 0, stream>>>(Wk, WkT, E, KVH * D);
    transpose_convert_kernel<<<dim3((KVH * D) / 32, E / 32), blk256, 0, stream>>>(Wv, WvT, E, KVH * D);
    transpose_convert_kernel<<<dim3(E / 32, (H * D) / 32), blk256, 0, stream>>>(Wo, WoT, H * D, E);

    // Projections (f32 A staged with convert; f16 BT direct from global)
    gemm_wmma_kernel<float, false><<<dim3((H * D) / 32, Mrow / 128), blk256, 0, stream>>>(
        query, WqT, Qp, nullptr, Mrow, H * D, E);
    gemm_wmma_kernel<float, false><<<dim3((KVH * D) / 32, Mrow / 128), blk256, 0, stream>>>(
        key, WkT, Kp, nullptr, Mrow, KVH * D, E);
    gemm_wmma_kernel<float, false><<<dim3((KVH * D) / 32, Mrow / 128), blk256, 0, stream>>>(
        value, WvT, Vp, nullptr, Mrow, KVH * D, E);

    // RoPE on Q and K
    rope_kernel<<<(Bn * S * (H + KVH) * 64) / 256, blk256, 0, stream>>>(Qp, Kp);

    // Causal GQA flash attention
    flash_attn_kernel<<<Bn * H * (S / 64), dim3(128), 0, stream>>>(Qp, Kp, Vp, AO);

    // Output projection (f16 A via Tensor Data Mover, f32 out)
    gemm_wmma_kernel<half_t, true><<<dim3(E / 32, Mrow / 128), blk256, 0, stream>>>(
        AO, WoT, nullptr, out, Mrow, E, H * D);
}
